// SPLICE_54434415509567
// MI455X (gfx1250) — matrix operand order repaired and checked by reading, not compiled, not run on hardware
//
#include <hip/hip_runtime.h>
#include <hip/hip_bf16.h>
#include <cstdint>
#include <cstddef>

// ---------------- problem constants (from reference) ----------------
#define RHO_F      5.0f
#define L1_F       0.01f
#define ADMM_ITERS 30
#define NS_ITERS   16          // Newton-Schulz iterations for M^{-1} (768x768 SPD)

static const int NIMG = 256;   // rows of image/text
static const int NC   = 8192;  // concepts
static const int DE   = 768;   // embedding dim

typedef __attribute__((ext_vector_type(2)))  float  v2f;
typedef __attribute__((ext_vector_type(8)))  float  v8f;
typedef __attribute__((ext_vector_type(4)))  int    v4i;
typedef __attribute__((ext_vector_type(8)))  __bf16 v8bf;
typedef __attribute__((ext_vector_type(16))) __bf16 v16bf;

typedef __attribute__((address_space(1))) v4i as1_v4i;
typedef __attribute__((address_space(3))) v4i as3_v4i;

// ---------------- CDNA5 async global->LDS helpers ----------------
__device__ __forceinline__ void cp_async16(const __bf16* g, __bf16* l) {
#if defined(__AMDGCN__) && __has_builtin(__builtin_amdgcn_global_load_async_to_lds_b128)
  // signature (from clang-22 diagnostic): (v4i AS1*, v4i AS3*, imm offset, imm cpol)
  __builtin_amdgcn_global_load_async_to_lds_b128(
      (as1_v4i*)(uintptr_t)g,                 // global address (AS1 == generic bits)
      (as3_v4i*)(unsigned)(uintptr_t)l,       // LDS offset = low 32 bits of generic
      /*offset=*/0, /*cpol=*/0);
#else
  *(v8bf*)l = *(const v8bf*)g;        // fallback (host pass / older toolchains)
#endif
}
__device__ __forceinline__ void async_wait0() {
#if defined(__AMDGCN__) && __has_builtin(__builtin_amdgcn_s_wait_asynccnt)
  __builtin_amdgcn_s_wait_asynccnt(0);
#elif defined(__AMDGCN__)
  asm volatile("s_wait_asynccnt 0x0" ::: "memory");
#endif
}

// =====================================================================
// bf16 WMMA GEMM, LDS-staged + double-buffered async copies.
//   C[M,N] = A[M,K](bf16 row-major) * B[K,N], B pre-packed Bp[k/32][n][32]
// Block: 256 thr = 8 wave32 waves; block tile 64x128; wave tile 32x32
// (2x2 fragments -> each A/B fragment feeds 2 WMMAs, 4 indep chains).
// Per BK=32 step: stage A(4KB)+B(8KB) -> LDS via async b128, dbl-buffered.
// M % 64 == 0, N % 128 == 0, K % 32 == 0 for all uses (256/768/8192).
// =====================================================================
__global__ __launch_bounds__(256)
void k_gemm_bf16(const __bf16* __restrict__ A, int lda,
                 const __bf16* __restrict__ Bp, int N,
                 float* __restrict__ Cf, __bf16* __restrict__ Cb,
                 int ldc, int K)
{
  __shared__ __bf16 As[2][64][32];     // [buf][row][k]    4KB each
  __shared__ __bf16 Bs[2][128 * 32];   // [buf][n*32+k]    8KB each

  const int tid  = threadIdx.x;
  const int lane = tid & 31;
  const int wave = tid >> 5;
  const int half = lane >> 4;          // K sub-range selector
  const int q    = lane & 15;          // row (A) / col (B,C) within 16-tile
  const int mblk = blockIdx.y * 64;
  const int nblk = blockIdx.x * 128;
  const int mb   = (wave & 1) * 32;    // wave tile origin in block tile
  const int nb   = (wave >> 1) * 32;

  // staging assignments (per thread): 1 A chunk + 2 B chunks of 16B
  const int ar   = tid >> 2;           // A row 0..63
  const int aseg = (tid & 3) * 8;      // A col segment 0/8/16/24
  const __bf16* agsrc = A + (size_t)(mblk + ar) * lda + aseg;
  const __bf16* bgsrc = Bp + (size_t)nblk * 32;   // + kt*N*32 per step

  const int nK = K >> 5;
  // prologue: stage k-tile 0 into buffer 0
  cp_async16(agsrc, &As[0][ar][aseg]);
  cp_async16(bgsrc + tid * 8,         &Bs[0][tid * 8]);
  cp_async16(bgsrc + (tid + 256) * 8, &Bs[0][(tid + 256) * 8]);

  v8f acc00 = {}, acc01 = {}, acc10 = {}, acc11 = {};

  for (int kt = 0; kt < nK; ++kt) {
    const int cur = kt & 1;
    async_wait0();
    __syncthreads();                    // stage `cur` visible to all waves

    if (kt + 1 < nK) {                  // stage next k-tile into other buffer
      const int nxt = cur ^ 1;
      cp_async16(agsrc + (kt + 1) * 32, &As[nxt][ar][aseg]);
      const __bf16* bs = bgsrc + (size_t)(kt + 1) * N * 32;
      cp_async16(bs + tid * 8,          &Bs[nxt][tid * 8]);
      cp_async16(bs + (tid + 256) * 8,  &Bs[nxt][(tid + 256) * 8]);
    }

    // fragments from LDS (ds_load_b128), fragment-native layouts
    v16bf a0, a1, b0, b1;
    {
      v8bf lo0 = *(const v8bf*)&As[cur][mb + q][16 * half];
      v8bf hi0 = *(const v8bf*)&As[cur][mb + q][16 * half + 8];
      v8bf lo1 = *(const v8bf*)&As[cur][mb + 16 + q][16 * half];
      v8bf hi1 = *(const v8bf*)&As[cur][mb + 16 + q][16 * half + 8];
#pragma unroll
      for (int i = 0; i < 8; ++i) {
        a0[i] = lo0[i]; a0[8 + i] = hi0[i];
        a1[i] = lo1[i]; a1[8 + i] = hi1[i];
      }
      b0 = *(const v16bf*)&Bs[cur][(nb + q) * 32 + 16 * half];
      b1 = *(const v16bf*)&Bs[cur][(nb + 16 + q) * 32 + 16 * half];
    }

    acc00 = __builtin_amdgcn_wmma_f32_16x16x32_bf16(false, a0, false, b0, (short)0, acc00, false, false);
    acc01 = __builtin_amdgcn_wmma_f32_16x16x32_bf16(false, a0, false, b1, (short)0, acc01, false, false);
    acc10 = __builtin_amdgcn_wmma_f32_16x16x32_bf16(false, a1, false, b0, (short)0, acc10, false, false);
    acc11 = __builtin_amdgcn_wmma_f32_16x16x32_bf16(false, a1, false, b1, (short)0, acc11, false, false);

    __syncthreads();                    // all reads of `cur` done before reuse
  }

  // store 2x2 fragment grid; C/D layout: VGPR r -> M = r + 8*half, N = q
#pragma unroll
  for (int mi = 0; mi < 2; ++mi) {
#pragma unroll
    for (int ni = 0; ni < 2; ++ni) {
      const v8f* acc = mi ? (ni ? &acc11 : &acc10) : (ni ? &acc01 : &acc00);
      const int col = nblk + nb + ni * 16 + q;
#pragma unroll
      for (int r = 0; r < 8; ++r) {
        const int row = mblk + mb + mi * 16 + r + 8 * half;
        if (Cf) Cf[(size_t)row * ldc + col] = (*acc)[r];
        else    Cb[(size_t)row * ldc + col] = (__bf16)((*acc)[r]);
      }
    }
  }
}

// =====================================================================
// f32 WMMA GEMM (V_WMMA_F32_16X16X4_F32), full precision, used for
// G = D^T D and the Newton-Schulz inverse iterations (768x768).
// atrans=1 reads A as stored [K,M] row-major (i.e. logical A = storedᵀ).
// =====================================================================
__global__ __launch_bounds__(256)
void k_gemm_f32(const float* __restrict__ A, int lda, int atrans,
                const float* __restrict__ B, int ldb,
                float* __restrict__ C, int ldc, int K)
{
  const int lane = threadIdx.x & 31;
  const int wave = threadIdx.x >> 5;
  const int half = lane >> 4;
  const int q    = lane & 15;
  const int m0   = blockIdx.y * 32 + (wave & 1) * 16;
  const int n0   = blockIdx.x * 64 + (wave >> 1) * 16;
  const int m = m0 + q, n = n0 + q;

  v8f acc = {};
  for (int kk = 0; kk < K; kk += 4) {
    const int k0 = kk + 2 * half;       // lane half selects K pair {0,1} vs {2,3}
    v2f a, b;
    if (atrans) {
      a.x = A[(size_t)k0 * lda + m];
      a.y = A[(size_t)(k0 + 1) * lda + m];
    } else {
      a = *(const v2f*)(A + (size_t)m * lda + k0);
    }
    b.x = B[(size_t)k0 * ldb + n];
    b.y = B[(size_t)(k0 + 1) * ldb + n];
    acc = __builtin_amdgcn_wmma_f32_16x16x4_f32(
        false, a, false, b, (short)0, acc, false, false);
  }
#pragma unroll
  for (int r = 0; r < 8; ++r)
    C[(size_t)(m0 + r + 8 * half) * ldc + (n0 + q)] = acc[r];
}

// ---------------- packing / conversion kernels ----------------
__global__ void k_conv_bf16(const float* __restrict__ s, __bf16* __restrict__ d, long n) {
  long i = (long)blockIdx.x * blockDim.x + threadIdx.x;
  const long st = (long)gridDim.x * blockDim.x;
  for (; i < n; i += st) d[i] = (__bf16)s[i];
}

// B = Sᵀ, S is [N,K] row-major (contiguous source): Bp[kt][n][kk]=S[n][32kt+kk]
__global__ void k_pack_contig(const __bf16* __restrict__ S, __bf16* __restrict__ Bp,
                              int N, long K) {
  const long nchunk = (long)N * K / 8;       // 8 bf16 per chunk
  long i = (long)blockIdx.x * blockDim.x + threadIdx.x;
  const long st = (long)gridDim.x * blockDim.x;
  for (; i < nchunk; i += st) {
    const int  c  = (int)(i & 3);
    const long t  = i >> 2;
    const int  n  = (int)(t % N);
    const long kt = t / N;
    *(v8bf*)(Bp + ((kt * N + n) << 5) + c * 8) =
        *(const v8bf*)(S + (size_t)n * K + kt * 32 + c * 8);
  }
}

// B = S, S is [K,N] row-major (strided gather): Bp[kt][n][kk]=S[32kt+kk][n]
__global__ void k_pack_strided(const __bf16* __restrict__ S, __bf16* __restrict__ Bp,
                               int N, long K) {
  const long total = (long)N * K;
  long i = (long)blockIdx.x * blockDim.x + threadIdx.x;
  const long st = (long)gridDim.x * blockDim.x;
  for (; i < total; i += st) {
    const int  kk = (int)(i & 31);
    const long t  = i >> 5;
    const int  n  = (int)(t % N);
    const long kt = t / N;
    Bp[i] = S[(kt * 32 + kk) * (long)N + n];
  }
}

// ---------------- elementwise / small kernels ----------------
__global__ void k_add_diag(float* M) {
  int i = blockIdx.x * blockDim.x + threadIdx.x;
  if (i < DE) M[(size_t)i * DE + i] += RHO_F;
}
__global__ void k_rowsum_abs(const float* __restrict__ M, float* __restrict__ rs) {
  __shared__ float red[256];
  const int i = blockIdx.x, t = threadIdx.x;
  float s = 0.f;
  for (int j = t; j < DE; j += 256) s += fabsf(M[(size_t)i * DE + j]);
  red[t] = s; __syncthreads();
  for (int k = 128; k > 0; k >>= 1) { if (t < k) red[t] += red[t + k]; __syncthreads(); }
  if (!t) rs[i] = red[0];
}
__global__ void k_maxinv(const float* __restrict__ rs, float* __restrict__ scal) {
  __shared__ float red[256];
  const int t = threadIdx.x;
  float m = 0.f;
  for (int i = t; i < DE; i += 256) m = fmaxf(m, rs[i]);
  red[t] = m; __syncthreads();
  for (int k = 128; k > 0; k >>= 1) { if (t < k) red[t] = fmaxf(red[t], red[t + k]); __syncthreads(); }
  if (!t) scal[0] = 1.0f / (red[0] * red[0]);   // X0 = M / ||M||^2
}
__global__ void k_scale_by(const float* __restrict__ s, const float* __restrict__ scal,
                           float* __restrict__ d, long n) {
  long i = (long)blockIdx.x * blockDim.x + threadIdx.x;
  const long st = (long)gridDim.x * blockDim.x;
  const float a = scal[0];
  for (; i < n; i += st) d[i] = s[i] * a;
}
__global__ void k_ns_t2(float* __restrict__ T, int n) {   // T <- 2I - T
  long i = (long)blockIdx.x * blockDim.x + threadIdx.x;
  const long st = (long)gridDim.x * blockDim.x, tot = (long)n * n;
  for (; i < tot; i += st) {
    const int r = (int)(i / n), c = (int)(i % n);
    T[i] = ((r == c) ? 2.0f : 0.0f) - T[i];
  }
}
__global__ void k_zero2(float* __restrict__ a, float* __restrict__ b, long n) {
  long i = (long)blockIdx.x * blockDim.x + threadIdx.x;
  const long st = (long)gridDim.x * blockDim.x;
  for (; i < n; i += st) { a[i] = 0.f; b[i] = 0.f; }
}
__global__ void k_ew_v(const float* __restrict__ Atb, const float* __restrict__ z,
                       const float* __restrict__ u, float* __restrict__ v,
                       __bf16* __restrict__ vbf, long n) {
  long i = (long)blockIdx.x * blockDim.x + threadIdx.x;
  const long st = (long)gridDim.x * blockDim.x;
  for (; i < n; i += st) {
    const float vv = Atb[i] + RHO_F * (z[i] - u[i]);
    v[i] = vv; vbf[i] = (__bf16)vv;
  }
}
__global__ void k_ew_zu(const float* __restrict__ v, const float* __restrict__ t3,
                        float* __restrict__ z, float* __restrict__ u,
                        __bf16* __restrict__ zbf, long n) {
  long i = (long)blockIdx.x * blockDim.x + threadIdx.x;
  const long st = (long)gridDim.x * blockDim.x;
  for (; i < n; i += st) {
    const float x  = (v[i] - t3[i]) * (1.0f / RHO_F);
    const float zn = fmaxf(x + u[i] - (L1_F / RHO_F), 0.0f);
    u[i] = u[i] + x - zn;
    z[i] = zn; zbf[i] = (__bf16)zn;
  }
}

// centered = normalize(normalize(img) - mean), emitted in bf16 (GEMM A operand)
__global__ __launch_bounds__(256)
void k_prep_centered(const float* __restrict__ img, const float* __restrict__ mean,
                     __bf16* __restrict__ out) {
  __shared__ float sx[DE]; __shared__ float red[256];
  const int r = blockIdx.x, t = threadIdx.x;
  const float* x = img + (size_t)r * DE;
  float loc = 0.f;
  for (int d = t; d < DE; d += 256) { const float v = x[d]; sx[d] = v; loc += v * v; }
  red[t] = loc; __syncthreads();
  for (int k = 128; k > 0; k >>= 1) { if (t < k) red[t] += red[t + k]; __syncthreads(); }
  const float n1 = fmaxf(sqrtf(red[0]), 1e-12f);
  __syncthreads();
  loc = 0.f;
  for (int d = t; d < DE; d += 256) { const float v = sx[d] / n1 - mean[d]; sx[d] = v; loc += v * v; }
  red[t] = loc; __syncthreads();
  for (int k = 128; k > 0; k >>= 1) { if (t < k) red[t] += red[t + k]; __syncthreads(); }
  const float n2 = fmaxf(sqrtf(red[0]), 1e-12f);
  for (int d = t; d < DE; d += 256) out[(size_t)r * DE + d] = (__bf16)(sx[d] / n2);
}

// recon = normalize(normalize(z@D) + mean) -> d_out (f32)
__global__ __launch_bounds__(256)
void k_finalize(const float* __restrict__ rec, const float* __restrict__ mean,
                float* __restrict__ out) {
  __shared__ float sx[DE]; __shared__ float red[256];
  const int r = blockIdx.x, t = threadIdx.x;
  const float* x = rec + (size_t)r * DE;
  float loc = 0.f;
  for (int d = t; d < DE; d += 256) { const float v = x[d]; sx[d] = v; loc += v * v; }
  red[t] = loc; __syncthreads();
  for (int k = 128; k > 0; k >>= 1) { if (t < k) red[t] += red[t + k]; __syncthreads(); }
  const float n1 = fmaxf(sqrtf(red[0]), 1e-12f);
  __syncthreads();
  loc = 0.f;
  for (int d = t; d < DE; d += 256) { const float v = sx[d] / n1 + mean[d]; sx[d] = v; loc += v * v; }
  red[t] = loc; __syncthreads();
  for (int k = 128; k > 0; k >>= 1) { if (t < k) red[t] += red[t + k]; __syncthreads(); }
  const float n2 = fmaxf(sqrtf(red[0]), 1e-12f);
  for (int d = t; d < DE; d += 256) out[(size_t)r * DE + d] = sx[d] / n2;
}

__global__ __launch_bounds__(256)
void k_norm_rows(const float* __restrict__ x, float* __restrict__ out) {
  __shared__ float red[256];
  const int r = blockIdx.x, t = threadIdx.x;
  const float* xr = x + (size_t)r * DE;
  float loc = 0.f;
  for (int d = t; d < DE; d += 256) { const float v = xr[d]; loc += v * v; }
  red[t] = loc; __syncthreads();
  for (int k = 128; k > 0; k >>= 1) { if (t < k) red[t] += red[t + k]; __syncthreads(); }
  const float n1 = fmaxf(sqrtf(red[0]), 1e-12f);
  for (int d = t; d < DE; d += 256) out[(size_t)r * DE + d] = xr[d] / n1;
}

// =====================================================================
// Host orchestration
// =====================================================================
static inline size_t al256(size_t x) { return (x + 255) & ~(size_t)255; }

extern "C" void kernel_launch(void* const* d_in, const int* in_sizes, int n_in,
                              void* d_out, int out_size, void* d_ws, size_t ws_size,
                              hipStream_t stream) {
  (void)in_sizes; (void)n_in; (void)out_size; (void)ws_size;
  const float* image = (const float*)d_in[0];   // [256,768]
  const float* text  = (const float*)d_in[1];   // [256,768]
  const float* mean  = (const float*)d_in[2];   // [768]
  const float* dict  = (const float*)d_in[3];   // [8192,768] f32
  float* out = (float*)d_out;                   // recon [256,768] ++ txt [256,768]

  // ---- workspace bump allocation (~127 MB) ----
  char* w = (char*)d_ws; size_t off = 0;
  auto alloc = [&](size_t bytes) { void* p = w + off; off = al256(off + bytes); return p; };
  const size_t NCDE = (size_t)NC * DE;          // 6291456
  const size_t NICC = (size_t)NIMG * NC;        // 2097152
  __bf16* Dbf     = (__bf16*)alloc(NCDE * 2);
  __bf16* Bp_Dt   = (__bf16*)alloc(NCDE * 2);   // packed B = Dᵀ  (K=768,  N=8192)
  __bf16* Bp_D    = (__bf16*)alloc(NCDE * 2);   // packed B = D   (K=8192, N=768)
  __bf16* Etb     = (__bf16*)alloc(NCDE * 2);   // Eᵀ = D @ M⁻¹   [8192,768]
  __bf16* Bp_E    = (__bf16*)alloc(NCDE * 2);   // packed B = E   (K=768,  N=8192)
  float*  Mmat    = (float*)alloc((size_t)DE * DE * 4);
  float*  X1      = (float*)alloc((size_t)DE * DE * 4);
  float*  X2      = (float*)alloc((size_t)DE * DE * 4);
  float*  T       = (float*)alloc((size_t)DE * DE * 4);
  __bf16* Minvb   = (__bf16*)alloc((size_t)DE * DE * 2);
  __bf16* Bp_Minv = (__bf16*)alloc((size_t)DE * DE * 2);
  __bf16* cenb    = (__bf16*)alloc((size_t)NIMG * DE * 2);
  float*  Atb     = (float*)alloc(NICC * 4);
  float*  zbuf    = (float*)alloc(NICC * 4);
  float*  ubuf    = (float*)alloc(NICC * 4);
  float*  vbuf    = (float*)alloc(NICC * 4);
  float*  t3      = (float*)alloc(NICC * 4);
  __bf16* vbf     = (__bf16*)alloc(NICC * 2);
  __bf16* zbf     = (__bf16*)alloc(NICC * 2);
  __bf16* t1b     = (__bf16*)alloc((size_t)NIMG * DE * 2);
  float*  rec0    = (float*)alloc((size_t)NIMG * DE * 4);
  float*  rs      = (float*)alloc(DE * 4);
  float*  scal    = (float*)alloc(256);

  const dim3 blk(256);
  const dim3 gs(2048);                                   // grid-stride kernels
  const dim3 gG(DE / 64, DE / 32);                       // f32 gemm 768x768
  const dim3 gEt(DE / 128, NC / 64);                     // bf16 gemm 8192x768
  const dim3 gWide(NC / 128, NIMG / 64);                 // bf16 gemm 256x8192
  const dim3 gThin(DE / 128, NIMG / 64);                 // bf16 gemm 256x768

  // ---- setup: dictionary in bf16 + packed layouts ----
  k_conv_bf16   <<<gs, blk, 0, stream>>>(dict, Dbf, (long)NCDE);
  k_pack_contig <<<gs, blk, 0, stream>>>(Dbf, Bp_Dt, NC, DE);   // B = Dᵀ
  k_pack_strided<<<gs, blk, 0, stream>>>(Dbf, Bp_D, DE, NC);    // B = D

  // ---- M = ρI + DᵀD in full f32 via f32-WMMA (A = Dᵀ read transposed) ----
  k_gemm_f32<<<gG, blk, 0, stream>>>(dict, DE, /*atrans=*/1, dict, DE, Mmat, DE, NC);
  k_add_diag<<<dim3(3), blk, 0, stream>>>(Mmat);

  // ---- M⁻¹ by Newton-Schulz (SPD, X0 = M/||M||², fixed iteration count) ----
  k_rowsum_abs<<<dim3(DE), blk, 0, stream>>>(Mmat, rs);
  k_maxinv    <<<dim3(1), blk, 0, stream>>>(rs, scal);
  k_scale_by  <<<gs, blk, 0, stream>>>(Mmat, scal, X1, (long)DE * DE);
  float* Xa = X1; float* Xb = X2;
  for (int it = 0; it < NS_ITERS; ++it) {
    k_gemm_f32<<<gG, blk, 0, stream>>>(Mmat, DE, 0, Xa, DE, T, DE, DE);   // T = M X
    k_ns_t2   <<<gs, blk, 0, stream>>>(T, DE);                            // T = 2I - T
    k_gemm_f32<<<gG, blk, 0, stream>>>(Xa, DE, 0, T, DE, Xb, DE, DE);     // X = X T
    float* tmp = Xa; Xa = Xb; Xb = tmp;
  }

  // ---- Eᵀ = D @ M⁻¹ (bf16), packed as B = E for the per-iter GEMM ----
  k_conv_bf16   <<<gs, blk, 0, stream>>>(Xa, Minvb, (long)DE * DE);
  k_pack_contig <<<gs, blk, 0, stream>>>(Minvb, Bp_Minv, DE, DE);          // M⁻¹ symmetric
  k_gemm_bf16   <<<gEt, blk, 0, stream>>>(Dbf, DE, Bp_Minv, DE, nullptr, Etb, DE, DE);
  k_pack_contig <<<gs, blk, 0, stream>>>(Etb, Bp_E, NC, DE);

  // ---- embeddings & Atb = centered @ Dᵀ ----
  k_prep_centered<<<dim3(NIMG), blk, 0, stream>>>(image, mean, cenb);
  k_gemm_bf16    <<<gWide, blk, 0, stream>>>(cenb, DE, Bp_Dt, NC, Atb, nullptr, NC, DE);
  k_zero2        <<<gs, blk, 0, stream>>>(zbuf, ubuf, (long)NICC);

  // ---- ADMM iterations (Woodbury form): x = (v - (v D) E)/ρ ----
  for (int it = 0; it < ADMM_ITERS; ++it) {
    k_ew_v     <<<gs, blk, 0, stream>>>(Atb, zbuf, ubuf, vbuf, vbf, (long)NICC);
    k_gemm_bf16<<<gThin, blk, 0, stream>>>(vbf, NC, Bp_D, DE, nullptr, t1b, DE, NC);   // t1 = v D
    k_gemm_bf16<<<gWide, blk, 0, stream>>>(t1b, DE, Bp_E, NC, t3, nullptr, NC, DE);    // t3 = t1 E
    k_ew_zu    <<<gs, blk, 0, stream>>>(vbuf, t3, zbuf, ubuf, zbf, (long)NICC);
  }

  // ---- recon = normalize(normalize(z @ D) + mean); txt = normalize(text) ----
  k_gemm_bf16<<<gThin, blk, 0, stream>>>(zbf, NC, Bp_D, DE, rec0, nullptr, DE, NC);
  k_finalize <<<dim3(NIMG), blk, 0, stream>>>(rec0, mean, out);
  k_norm_rows<<<dim3(NIMG), blk, 0, stream>>>(text, out + (size_t)NIMG * DE);
}
